// tfESN_37580963840555
// MI455X (gfx1250) — matrix operand order repaired and checked by reading, not compiled, not run on hardware
//
#include <hip/hip_runtime.h>
#include <hip/hip_bf16.h>
#include <math.h>

#define N_RES 4096
#define BATCH 64
#define N_IN  128
#define N_OUT 128

// K is processed in 32-wide blocks: 4 (u) + 128 (x) + 4 (y) = 136 blocks
#define KB_U 4
#define KB_X 128
#define KB_Y 4
#define KB_TOT 136

typedef __attribute__((ext_vector_type(16))) __bf16 v16bf;
typedef __attribute__((ext_vector_type(8)))  float  v8f;
typedef __attribute__((ext_vector_type(4)))  float  f32x4;

union Pack4 { __bf16 b[4]; unsigned long long u; };

// D = A(16x32 bf16) x B(32x16 bf16) + C(16x16 f32), wave32
static __device__ inline v8f wmma_bf16(v16bf a, v16bf b, v8f c) {
  return __builtin_amdgcn_wmma_f32_16x16x32_bf16(
      /*neg_a=*/false, a, /*neg_b=*/false, b,
      /*c_mod=*/(short)0, c, /*reuse_a=*/false, /*reuse_b=*/false);
}

// A operand per-lane chunks for the 16-bit 16x32 layout: k in
// [h*8, h*8+8) and [h*8+16, h*8+24); p points at row + kb + half*8.
static __device__ inline void load_split_a16(const float* __restrict__ p,
                                             v16bf& hi, v16bf& lo) {
  const f32x4* q0 = (const f32x4*)p;
  const f32x4* q1 = (const f32x4*)(p + 16);
#pragma unroll
  for (int c = 0; c < 2; ++c) {
    f32x4 v = q0[c];
#pragma unroll
    for (int e = 0; e < 4; ++e) {
      float f = v[e];
      __bf16 h = (__bf16)f;
      hi[c * 4 + e] = h;
      lo[c * 4 + e] = (__bf16)(f - (float)h);
    }
  }
#pragma unroll
  for (int c = 0; c < 2; ++c) {
    f32x4 v = q1[c];
#pragma unroll
    for (int e = 0; e < 4; ++e) {
      float f = v[e];
      __bf16 h = (__bf16)f;
      hi[8 + c * 4 + e] = h;
      lo[8 + c * 4 + e] = (__bf16)(f - (float)h);
    }
  }
}

// ---------------------------------------------------------------------------
// Pass 1: split A = [u | x | y^T] (64 x 4352 f32) into bf16 hi/lo,
// pre-swizzled into the WMMA A-operand layout.
// ws layout: chunk(kblock, btile, lane) at ((kblock*4+btile)*32+lane)*64 B:
//   bytes [0,32) = 16 bf16 hi, bytes [32,64) = 16 bf16 lo.
// Total: 136*4*32*64 = 1,114,112 bytes.
// ---------------------------------------------------------------------------
__global__ __launch_bounds__(128, 1) void esn_pack_a(
    const float* __restrict__ U,   // [64, 128]
    const float* __restrict__ Y,   // [128, 64]
    const float* __restrict__ X,   // [64, 4096]
    char* __restrict__ Aws)
{
  const int kblock = blockIdx.x;        // 0..135
  const int t      = threadIdx.x;       // 0..127
  const int btile  = t >> 5;
  const int lane   = t & 31;
  const int m      = lane & 15;
  const int half   = lane >> 4;
  const int b      = btile * 16 + m;

  v16bf hi, lo;
  if (kblock < KB_U) {
    load_split_a16(U + (size_t)b * N_IN + kblock * 32 + half * 8, hi, lo);
  } else if (kblock < KB_U + KB_X) {
    load_split_a16(X + (size_t)b * N_RES + (kblock - KB_U) * 32 + half * 8, hi, lo);
  } else {
    const int ko = (kblock - (KB_U + KB_X)) * 32;
#pragma unroll
    for (int j = 0; j < 16; ++j) {
      int o = ko + half * 8 + (j < 8 ? j : j + 8);
      float f = Y[(size_t)o * BATCH + b];     // A_y[b, o] = y_last[o, b]
      __bf16 h = (__bf16)f;
      hi[j] = h;
      lo[j] = (__bf16)(f - (float)h);
    }
  }

  char* dst = Aws + ((size_t)(kblock * 4 + btile) * 32 + lane) * 64;
  *(v16bf*)dst        = hi;
  *(v16bf*)(dst + 32) = lo;
}

// ---------------------------------------------------------------------------
// Pass 2: GEMM + bias + tanh. Block = 128 threads = 4 waves; blockIdx.x picks
// 16 output columns (weight rows), wave id picks 16 batch rows. Weight tiles
// are loaded+split ONCE per block into LDS (ping-pong), shared by all 4 waves
// via ds_load; A operands come pre-split from the workspace.
// Each segment is a separate inlined loop with a direct global pointer so the
// weight stream stays on global_load_b128 (no FLAT fallback / DScnt coupling).
// ---------------------------------------------------------------------------
__global__ __launch_bounds__(128, 1) void esn_step_wmma(
    const char*  __restrict__ Aws,
    const float* __restrict__ Win,   // [4096, 128]
    const float* __restrict__ Wres,  // [4096, 4096]
    const float* __restrict__ Wout,  // [4096, 128]
    const float* __restrict__ Bias,  // [4096]
    float* __restrict__ Out)         // [64, 4096]
{
  // [buf][lane][0..15]=hi, [16..31]=lo ; 4 KB total
  __shared__ __attribute__((aligned(64))) __bf16 lds[2][32][32];

  const int t    = threadIdx.x;
  const int lane = t & 31;
  const int wave = t >> 5;          // = batch tile
  const int nl   = lane & 15;
  const int half = lane >> 4;
  const int nbase = blockIdx.x << 4;
  const int bbase = wave << 4;

  // writer-side mapping: thread t stages 4 consecutive k of weight row wr_row
  const int wr_row  = t >> 3;                        // 0..15
  const int wr_k    = (t & 7) * 4;                   // 0,4,...,28
  const int wr_lane = wr_row + (((wr_k >> 3) & 1) << 4);   // dest operand lane
  const int wr_j    = (wr_k & 7) + ((wr_k >> 4) << 3);     // element offset in chunk

  v8f acc = {0.f, 0.f, 0.f, 0.f, 0.f, 0.f, 0.f, 0.f};
  int kidx = 0;

  // Inlined per-segment loop; wsrc is a direct GEP off a kernel arg, so the
  // stage load keeps the global address space.
  auto run_seg = [&](const float* __restrict__ wsrc, int nkb_c, bool pref) {
#pragma unroll 2
    for (int i = 0; i < nkb_c; ++i, ++kidx) {
      const int buf = kidx & 1;

      // --- cooperative stage: load 4 f32, split, pack 2x8B into LDS ---
      {
        const float* p = wsrc + i * 32;
        if (pref) __builtin_prefetch(p + 512, 0, 0);  // ~16 iters ahead
        f32x4 v = *(const f32x4*)p;
        Pack4 h, l;
#pragma unroll
        for (int e = 0; e < 4; ++e) {
          float f = v[e];
          __bf16 hh = (__bf16)f;
          h.b[e] = hh;
          l.b[e] = (__bf16)(f - (float)hh);
        }
        *(unsigned long long*)&lds[buf][wr_lane][wr_j]      = h.u;
        *(unsigned long long*)&lds[buf][wr_lane][16 + wr_j] = l.u;
      }
      __syncthreads();   // ping-pong: one barrier per iteration suffices

      // --- operands ---
      const char* ap = Aws + ((size_t)(kidx * 4 + wave) * 32 + lane) * 64;
      v16bf ah = *(const v16bf*)ap;
      v16bf al = *(const v16bf*)(ap + 32);
      v16bf bh = *(const v16bf*)&lds[buf][lane][0];
      v16bf bl = *(const v16bf*)&lds[buf][lane][16];

      // bf16x3: a_lo*b_hi + a_hi*b_lo + a_hi*b_hi, f32 accumulate
      acc = wmma_bf16(al, bh, acc);
      acc = wmma_bf16(ah, bl, acc);
      acc = wmma_bf16(ah, bh, acc);
    }
  };

  run_seg(Win  + (size_t)(nbase + wr_row) * N_IN  + wr_k, KB_U, false);
  run_seg(Wres + (size_t)(nbase + wr_row) * N_RES + wr_k, KB_X, true);
  run_seg(Wout + (size_t)(nbase + wr_row) * N_OUT + wr_k, KB_Y, false);

  // epilogue: + bias, tanh, store (C layout: lane<16 -> M=v, else M=v+8)
  const float bn = Bias[nbase + nl];
#pragma unroll
  for (int v = 0; v < 8; ++v) {
    int m = (half << 3) + v;
    Out[(size_t)(bbase + m) * N_RES + (nbase + nl)] = tanhf(acc[v] + bn);
  }
}

extern "C" void kernel_launch(void* const* d_in, const int* in_sizes, int n_in,
                              void* d_out, int out_size, void* d_ws, size_t ws_size,
                              hipStream_t stream) {
  (void)in_sizes; (void)n_in; (void)out_size; (void)ws_size;
  const float* U    = (const float*)d_in[0];
  const float* Y    = (const float*)d_in[1];
  const float* X    = (const float*)d_in[2];
  const float* Win  = (const float*)d_in[3];
  const float* Wres = (const float*)d_in[4];
  const float* Wout = (const float*)d_in[5];
  const float* Bias = (const float*)d_in[6];
  float* Out = (float*)d_out;
  char* Aws  = (char*)d_ws;   // needs 1,114,112 bytes

  esn_pack_a<<<dim3(KB_TOT), dim3(128), 0, stream>>>(U, Y, X, Aws);
  esn_step_wmma<<<dim3(N_RES / 16), dim3(128), 0, stream>>>(
      Aws, Win, Wres, Wout, Bias, Out);
}
